// Gemma4VisionEncoderLayer_28802050687680
// MI455X (gfx1250) — compile-verified, matrix operand order checked by reading, not run on hardware
//
#include <hip/hip_runtime.h>
#include <hip/hip_bf16.h>
#include <math.h>

#define T_TOK   2048
#define L_SEQ   1024
#define D_MODEL 1152
#define NHEAD   16
#define HDIM    72
#define FF_DIM  4304
#define NLAYER  4
#define PDIM    768
#define GRID_W  32
#define EPSF    1e-6f

typedef __attribute__((ext_vector_type(16))) __bf16 v16bf;
typedef __attribute__((ext_vector_type(8)))  __bf16 v8bf;
typedef __attribute__((ext_vector_type(4)))  __bf16 v4bf;
typedef __attribute__((ext_vector_type(8)))  float  v8f;

__device__ inline __bf16 f2bf(float f) { return (__bf16)f; }   // native v_cvt on gfx1250

// Load a 16x32 bf16 operand fragment from an LDS tile (row stride ldk halfwords).
// CDNA5 16-bit A/B layout: lane (r = lane&15, hi = lane>>4) holds
// K = hi*8 + 0..7 in elements 0..7 and K = 16 + hi*8 + 0..7 in elements 8..15.
__device__ inline v16bf load_frag(const __bf16* p0, int ldk) {
  const int lane = threadIdx.x & 31;
  const int r  = lane & 15;
  const int hi = lane >> 4;
  const __bf16* p = p0 + r * ldk + hi * 8;
  v8bf lo = *(const v8bf*)p;
  v8bf hh = *(const v8bf*)(p + 16);
  v16bf out;
#pragma unroll
  for (int i = 0; i < 8; ++i) { out[i] = lo[i]; out[i + 8] = hh[i]; }
  return out;
}

// ---------------------------------------------------------------------------
// GEMM: C[M,N] = A[M,K] * B[K,N]; fp32 in/out, bf16 WMMA core, f32 accumulate.
// Block 256 threads = 8 waves (4 down M x 2 across N); tile 128x128; K step 32.
// Preconditions: M%128==0, K%4==0, N%4==0.
// EDGE=false additionally assumes N%128==0 and K%32==0 (no guards at all).
// ---------------------------------------------------------------------------
template <bool EDGE>
__global__ __launch_bounds__(256)
void gemm_bf16_wmma(const float* __restrict__ A, const float* __restrict__ B,
                    float* __restrict__ C, int M, int N, int K)
{
  const int tid  = threadIdx.x;
  const int w    = tid >> 5;
  const int lane = tid & 31;
  const int n16  = lane & 15;
  const int hi   = lane >> 4;
  const int wm   = w & 3;            // 4 waves down M (32 rows each)
  const int wn   = w >> 2;           // 2 waves across N (64 cols each)
  const int m0   = blockIdx.y * 128;
  const int n0   = blockIdx.x * 128;

  __shared__ alignas(16) __bf16 As[128][40];    // +8 halfword pad vs bank conflicts
  __shared__ alignas(16) __bf16 Bts[128][40];   // B tile transposed (n-major)

  v8f acc[2][4];
#pragma unroll
  for (int i = 0; i < 2; ++i)
#pragma unroll
    for (int t = 0; t < 4; ++t) acc[i][t] = (v8f){};

  const int arow = tid >> 3;         // 0..31
  const int ac4  = (tid & 7) * 4;    // 0..28
  const int bk   = tid >> 5;         // 0..7
  const int bn4  = (tid & 31) * 4;   // 0..124

  for (int k0 = 0; k0 < K; k0 += 32) {
    // ---- stage A tile 128x32 (f32 -> bf16, packed 8B LDS stores) ----
#pragma unroll
    for (int rr = 0; rr < 4; ++rr) {
      const int r = arow + rr * 32;
      float4 t = {0.f, 0.f, 0.f, 0.f};
      if (!EDGE || (k0 + ac4 + 4 <= K))
        t = *(const float4*)(A + (size_t)(m0 + r) * K + k0 + ac4);
      v4bf pk = { f2bf(t.x), f2bf(t.y), f2bf(t.z), f2bf(t.w) };
      *(v4bf*)&As[r][ac4] = pk;
    }
    // ---- stage B tile 32x128, transposed into Bts[n][k] ----
#pragma unroll
    for (int pp = 0; pp < 4; ++pp) {
      const int kk = bk + pp * 8;
      float4 t = {0.f, 0.f, 0.f, 0.f};
      if (!EDGE || ((k0 + kk < K) && (n0 + bn4 + 4 <= N)))
        t = *(const float4*)(B + (size_t)(k0 + kk) * N + n0 + bn4);
      Bts[bn4 + 0][kk] = f2bf(t.x);
      Bts[bn4 + 1][kk] = f2bf(t.y);
      Bts[bn4 + 2][kk] = f2bf(t.z);
      Bts[bn4 + 3][kk] = f2bf(t.w);
    }
    // prefetch next k-slab
    if (k0 + 32 < K) {
      __builtin_prefetch(A + (size_t)(m0 + arow) * K + k0 + 32, 0, 1);
      if (!EDGE || (k0 + 32 + bk < K))
        __builtin_prefetch(B + (size_t)(k0 + 32 + bk) * N + n0 + bn4, 0, 1);
    }
    __syncthreads();

    v16bf a0 = load_frag(&As[wm * 32][0],      40);
    v16bf a1 = load_frag(&As[wm * 32 + 16][0], 40);
#pragma unroll
    for (int t = 0; t < 4; ++t) {
      v16bf bf = load_frag(&Bts[wn * 64 + t * 16][0], 40);
      acc[0][t] = __builtin_amdgcn_wmma_f32_16x16x32_bf16(false, a0, false, bf, (short)0, acc[0][t], false, false);
      acc[1][t] = __builtin_amdgcn_wmma_f32_16x16x32_bf16(false, a1, false, bf, (short)0, acc[1][t], false, false);
    }
    __syncthreads();
  }

  // ---- store: C layout lane holds col n16, rows r + 8*hi ----
  const int cm = m0 + wm * 32;
  const int cn = n0 + wn * 64;
#pragma unroll
  for (int i = 0; i < 2; ++i)
#pragma unroll
    for (int t = 0; t < 4; ++t) {
      const int nc = cn + t * 16 + n16;
      if (!EDGE || nc < N) {
#pragma unroll
        for (int r = 0; r < 8; ++r) {
          const int mr = cm + i * 16 + r + 8 * hi;
          C[(size_t)mr * N + nc] = acc[i][t][r];
        }
      }
    }
}

// ---------------------------------------------------------------------------
// Flash attention with WMMA. Block = 256 threads (8 waves x 16 queries) for one
// (batch, head). HD=72 padded to 96 for the QK^T K-dim, to 80 for PV N-tiles.
// ---------------------------------------------------------------------------
__global__ __launch_bounds__(256)
void flash_attn_kernel(const float* __restrict__ Q, const float* __restrict__ Kb,
                       const float* __restrict__ Vb, float* __restrict__ O)
{
  const int tid  = threadIdx.x;
  const int w    = tid >> 5;
  const int lane = tid & 31;
  const int n16  = lane & 15;
  const int hi   = lane >> 4;
  const int bh   = blockIdx.y;
  const int b    = bh / NHEAD;
  const int h    = bh % NHEAD;
  const int q0   = blockIdx.x * 128;
  const float scaleF = rsqrtf((float)HDIM);

  __shared__ alignas(16) __bf16 Kt[32][104];   // 32 keys x 96(pad hd) (+8 pad)
  __shared__ alignas(16) __bf16 Vt[80][40];    // transposed V: 80(pad hd) x 32 keys
  __shared__ alignas(16) __bf16 Pt[8][16][40]; // per-wave probs 16q x 32keys

  // Q fragments for this wave's 16 queries (3 K-chunks of 32, hd padded)
  v16bf qf[3];
  {
    const int row = b * L_SEQ + q0 + w * 16 + n16;
    const float* qp = Q + (size_t)row * D_MODEL + h * HDIM;
#pragma unroll
    for (int kc = 0; kc < 3; ++kc) {
      const int k0e = kc * 32 + hi * 8;
      v16bf f;
#pragma unroll
      for (int j = 0; j < 8; ++j) {
        const int kA = k0e + j, kB = k0e + 16 + j;
        f[j]     = (kA < HDIM) ? f2bf(qp[kA]) : f2bf(0.f);
        f[j + 8] = (kB < HDIM) ? f2bf(qp[kB]) : f2bf(0.f);
      }
      qf[kc] = f;
    }
  }

  v8f o[5];
  float m[8], l[8];
#pragma unroll
  for (int t = 0; t < 5; ++t) o[t] = (v8f){};
#pragma unroll
  for (int r = 0; r < 8; ++r) { m[r] = -1e30f; l[r] = 0.f; }

  for (int s0 = 0; s0 < L_SEQ; s0 += 32) {
    // stage K tile (row = key, col = hd padded to 96)
    for (int idx = tid; idx < 32 * 96; idx += 256) {
      const int key = idx / 96, cc = idx % 96;
      const float v = (cc < HDIM)
          ? Kb[(size_t)(b * L_SEQ + s0 + key) * D_MODEL + h * HDIM + cc] : 0.f;
      Kt[key][cc] = f2bf(v);
    }
    // stage V tile transposed (row = hd padded to 80, col = key)
    for (int idx = tid; idx < 32 * 80; idx += 256) {
      const int key = idx / 80, hd = idx % 80;
      const float v = (hd < HDIM)
          ? Vb[(size_t)(b * L_SEQ + s0 + key) * D_MODEL + h * HDIM + hd] : 0.f;
      Vt[hd][key] = f2bf(v);
    }
    __syncthreads();

    // S = Q * K^T for 16 queries x 32 keys (2 N-tiles, 3 K-chunks)
    v8f s0v = {}, s1v = {};
#pragma unroll
    for (int kc = 0; kc < 3; ++kc) {
      v16bf bk0 = load_frag(&Kt[0][kc * 32],  104);
      v16bf bk1 = load_frag(&Kt[16][kc * 32], 104);
      s0v = __builtin_amdgcn_wmma_f32_16x16x32_bf16(false, qf[kc], false, bk0, (short)0, s0v, false, false);
      s1v = __builtin_amdgcn_wmma_f32_16x16x32_bf16(false, qf[kc], false, bk1, (short)0, s1v, false, false);
    }

    // streaming softmax update (rows live across 16-lane halves)
    float alpha[8];
#pragma unroll
    for (int r = 0; r < 8; ++r) {
      float a  = s0v[r] * scaleF;
      float bb = s1v[r] * scaleF;
      float mx = fmaxf(a, bb);
      mx = fmaxf(mx, __shfl_xor(mx, 1, 16));
      mx = fmaxf(mx, __shfl_xor(mx, 2, 16));
      mx = fmaxf(mx, __shfl_xor(mx, 4, 16));
      mx = fmaxf(mx, __shfl_xor(mx, 8, 16));
      const float mnew = fmaxf(m[r], mx);
      alpha[r] = __expf(m[r] - mnew);
      const float p0 = __expf(a - mnew);
      const float p1 = __expf(bb - mnew);
      s0v[r] = p0; s1v[r] = p1;
      float rs = p0 + p1;
      rs += __shfl_xor(rs, 1, 16);
      rs += __shfl_xor(rs, 2, 16);
      rs += __shfl_xor(rs, 4, 16);
      rs += __shfl_xor(rs, 8, 16);
      l[r] = l[r] * alpha[r] + rs;
      m[r] = mnew;
    }
#pragma unroll
    for (int t = 0; t < 5; ++t)
#pragma unroll
      for (int r = 0; r < 8; ++r) o[t][r] *= alpha[r];

    // spill P (C-layout) into per-wave LDS, reload as A-fragment
#pragma unroll
    for (int r = 0; r < 8; ++r) {
      Pt[w][r + 8 * hi][n16]      = f2bf(s0v[r]);
      Pt[w][r + 8 * hi][16 + n16] = f2bf(s1v[r]);
    }
    asm volatile("s_wait_dscnt 0" ::: "memory");
    v16bf pf = load_frag(&Pt[w][0][0], 40);
#pragma unroll
    for (int t = 0; t < 5; ++t) {
      v16bf bv = load_frag(&Vt[t * 16][0], 40);
      o[t] = __builtin_amdgcn_wmma_f32_16x16x32_bf16(false, pf, false, bv, (short)0, o[t], false, false);
    }
    __syncthreads();
  }

  // normalize and store (skip hd pad columns)
  const int row = b * L_SEQ + q0 + w * 16;
#pragma unroll
  for (int t = 0; t < 5; ++t) {
    const int hd = t * 16 + n16;
    if (hd < HDIM) {
#pragma unroll
      for (int r = 0; r < 8; ++r) {
        const int mrow = r + 8 * hi;
        O[(size_t)(row + mrow) * D_MODEL + h * HDIM + hd] = o[t][r] / l[r];
      }
    }
  }
}

// ---------------------------------------------------------------------------
// Elementwise / normalization kernels
// ---------------------------------------------------------------------------
__global__ __launch_bounds__(256)
void prep_patches_kernel(const float* __restrict__ inputs, float* __restrict__ P)
{
  const int i = blockIdx.x * 256 + threadIdx.x;
  if (i >= T_TOK * PDIM) return;
  const int token = i / PDIM, e = i % PDIM;
  const int b = token / L_SEQ, s = token % L_SEQ;
  const int gy = s / GRID_W, gx = s % GRID_W;
  const int pe = e / 3, c = e % 3;
  const int py = pe / 16, px = pe % 16;
  const size_t src = (((size_t)b * 512 + gy * 16 + py) * 512 + gx * 16 + px) * 3 + c;
  P[i] = 2.f * inputs[src] - 1.f;
}

__global__ __launch_bounds__(256)
void add_posemb_kernel(float* __restrict__ X, const float* __restrict__ posemb)
{
  const int i = blockIdx.x * 256 + threadIdx.x;
  if (i >= T_TOK * D_MODEL) return;
  const int token = i / D_MODEL, d = i % D_MODEL;
  const int s = token % L_SEQ;
  const int gy = s / GRID_W, gx = s % GRID_W;
  X[i] += posemb[(size_t)gx * 2 * D_MODEL + d] + posemb[(size_t)gy * 2 * D_MODEL + D_MODEL + d];
}

__global__ __launch_bounds__(256)
void rmsnorm_kernel(const float* __restrict__ X, const float* __restrict__ sc,
                    float* __restrict__ Y)
{
  const int row = blockIdx.x;
  const int tid = threadIdx.x;
  const float* xr = X + (size_t)row * D_MODEL;
  float ss = 0.f;
  for (int e = tid; e < D_MODEL; e += 256) { const float v = xr[e]; ss += v * v; }
  __shared__ float red[8];
  for (int off = 16; off > 0; off >>= 1) ss += __shfl_xor(ss, off, 32);
  if ((tid & 31) == 0) red[tid >> 5] = ss;
  __syncthreads();
  float tot = 0.f;
#pragma unroll
  for (int i = 0; i < 8; ++i) tot += red[i];
  const float inv = rsqrtf(tot / (float)D_MODEL + EPSF);
  for (int e = tid; e < D_MODEL; e += 256)
    Y[(size_t)row * D_MODEL + e] = xr[e] * inv * (1.f + sc[e]);
}

__global__ __launch_bounds__(256)
void resadd_rms_kernel(float* __restrict__ X, const float* __restrict__ Y,
                       const float* __restrict__ sc)
{
  const int row = blockIdx.x;
  const int tid = threadIdx.x;
  const float* yr = Y + (size_t)row * D_MODEL;
  float ss = 0.f;
  for (int e = tid; e < D_MODEL; e += 256) { const float v = yr[e]; ss += v * v; }
  __shared__ float red[8];
  for (int off = 16; off > 0; off >>= 1) ss += __shfl_xor(ss, off, 32);
  if ((tid & 31) == 0) red[tid >> 5] = ss;
  __syncthreads();
  float tot = 0.f;
#pragma unroll
  for (int i = 0; i < 8; ++i) tot += red[i];
  const float inv = rsqrtf(tot / (float)D_MODEL + EPSF);
  for (int e = tid; e < D_MODEL; e += 256)
    X[(size_t)row * D_MODEL + e] += yr[e] * inv * (1.f + sc[e]);
}

// per (token, head): RMSNorm over HD=72 with (1+scale), then factorized 2D RoPE
__global__ __launch_bounds__(96)
void headnorm_rope_kernel(float* __restrict__ X, const float* __restrict__ sc, int do_rope)
{
  const int blk = blockIdx.x;        // token*NHEAD + h
  const int token = blk / NHEAD, h = blk % NHEAD;
  const int tid = threadIdx.x;
  float* base = X + (size_t)token * D_MODEL + h * HDIM;

  __shared__ float s[HDIM];
  __shared__ float red[3];
  const float v = (tid < HDIM) ? base[tid] : 0.f;
  float ss = v * v;
  for (int off = 16; off > 0; off >>= 1) ss += __shfl_xor(ss, off, 32);
  if ((tid & 31) == 0) red[tid >> 5] = ss;
  __syncthreads();
  const float tot = red[0] + red[1] + red[2];
  const float inv = rsqrtf(tot / (float)HDIM + EPSF);
  if (tid < HDIM) s[tid] = v * inv * (1.f + sc[tid]);
  __syncthreads();

  if (!do_rope) {
    if (tid < HDIM) base[tid] = s[tid];
    return;
  }
  const int sidx = token % L_SEQ;
  const float px = (float)(sidx % GRID_W);
  const float py = (float)(sidx / GRID_W);
  if (tid < 36) {
    const int part = tid / 18;       // 0: x-axis dims [0,36), 1: y-axis dims [36,72)
    const int j = tid % 18;
    const float pos = part ? py : px;
    const float ts = __powf(100.0f, (float)j / 18.0f);
    const float inp = pos / ts;
    const float sn = __sinf(inp), cs = __cosf(inp);
    const int o = part * 36;
    const float a = s[o + j], bb = s[o + j + 18];
    base[o + j]      = a * cs - bb * sn;
    base[o + j + 18] = bb * cs + a * sn;
  }
}

__global__ __launch_bounds__(256)
void geglu_kernel(float* __restrict__ G, const float* __restrict__ U)
{
  const int i = blockIdx.x * 256 + threadIdx.x;
  if (i >= T_TOK * FF_DIM) return;
  const float g = G[i];
  const float t = 0.7978845608028654f * (g + 0.044715f * g * g * g);
  G[i] = 0.5f * g * (1.f + tanhf(t)) * U[i];
}

__global__ __launch_bounds__(256)
void finalize_kernel(const float* __restrict__ X, const float* __restrict__ bias,
                     const float* __restrict__ scale, float* __restrict__ out)
{
  const int i = blockIdx.x * 256 + threadIdx.x;
  if (i >= T_TOK * D_MODEL) return;
  const int d = i % D_MODEL;
  out[i] = (X[i] * sqrtf((float)D_MODEL) - bias[d]) * scale[d];
}

// ---------------------------------------------------------------------------
// Orchestration
// ---------------------------------------------------------------------------
extern "C" void kernel_launch(void* const* d_in, const int* in_sizes, int n_in,
                              void* d_out, int out_size, void* d_ws, size_t ws_size,
                              hipStream_t stream)
{
  (void)in_sizes; (void)n_in; (void)out_size; (void)ws_size;
  const float* inputs    = (const float*)d_in[0];
  const float* Wp        = (const float*)d_in[1];
  const float* posemb    = (const float*)d_in[2];
  const float* Wq        = (const float*)d_in[3];
  const float* Wk        = (const float*)d_in[4];
  const float* Wv        = (const float*)d_in[5];
  const float* Wo        = (const float*)d_in[6];
  const float* Wg        = (const float*)d_in[7];
  const float* Wu        = (const float*)d_in[8];
  const float* Wd        = (const float*)d_in[9];
  const float* qn        = (const float*)d_in[10];
  const float* kn        = (const float*)d_in[11];
  const float* vn        = (const float*)d_in[12];
  const float* pre_attn  = (const float*)d_in[13];
  const float* post_attn = (const float*)d_in[14];
  const float* pre_ffw   = (const float*)d_in[15];
  const float* post_ffw  = (const float*)d_in[16];
  const float* std_bias  = (const float*)d_in[17];
  const float* std_scale = (const float*)d_in[18];
  float* out = (float*)d_out;

  float* ws = (float*)d_ws;
  size_t off = 0;
  float* x       = ws + off; off += (size_t)T_TOK * D_MODEL;
  float* hx      = ws + off; off += (size_t)T_TOK * D_MODEL;
  float* qb      = ws + off; off += (size_t)T_TOK * D_MODEL;
  float* kb      = ws + off; off += (size_t)T_TOK * D_MODEL;
  float* vb      = ws + off; off += (size_t)T_TOK * D_MODEL;
  float* attnb   = ws + off; off += (size_t)T_TOK * D_MODEL;
  float* g       = ws + off; off += (size_t)T_TOK * FF_DIM;
  float* u       = ws + off; off += (size_t)T_TOK * FF_DIM;
  float* patches = ws + off; off += (size_t)T_TOK * PDIM;

  auto gemm = [&](const float* Ap, const float* Bp, float* Cp, int M, int N, int K) {
    dim3 grid((N + 127) / 128, (M + 127) / 128);
    if ((N % 128 == 0) && (K % 32 == 0))
      gemm_bf16_wmma<false><<<grid, 256, 0, stream>>>(Ap, Bp, Cp, M, N, K);
    else
      gemm_bf16_wmma<true><<<grid, 256, 0, stream>>>(Ap, Bp, Cp, M, N, K);
  };

  // entry: patchify -> embed -> posemb
  prep_patches_kernel<<<(T_TOK * PDIM + 255) / 256, 256, 0, stream>>>(inputs, patches);
  gemm(patches, Wp, x, T_TOK, D_MODEL, PDIM);
  add_posemb_kernel<<<(T_TOK * D_MODEL + 255) / 256, 256, 0, stream>>>(x, posemb);

  for (int i = 0; i < NLAYER; ++i) {
    // attention sub-block
    rmsnorm_kernel<<<T_TOK, 256, 0, stream>>>(x, pre_attn + (size_t)i * D_MODEL, hx);
    gemm(hx, Wq + (size_t)i * D_MODEL * D_MODEL, qb, T_TOK, D_MODEL, D_MODEL);
    gemm(hx, Wk + (size_t)i * D_MODEL * D_MODEL, kb, T_TOK, D_MODEL, D_MODEL);
    gemm(hx, Wv + (size_t)i * D_MODEL * D_MODEL, vb, T_TOK, D_MODEL, D_MODEL);
    headnorm_rope_kernel<<<T_TOK * NHEAD, 96, 0, stream>>>(qb, qn + (size_t)i * HDIM, 1);
    headnorm_rope_kernel<<<T_TOK * NHEAD, 96, 0, stream>>>(kb, kn + (size_t)i * HDIM, 1);
    headnorm_rope_kernel<<<T_TOK * NHEAD, 96, 0, stream>>>(vb, vn + (size_t)i * HDIM, 0);
    flash_attn_kernel<<<dim3(L_SEQ / 128, 2 * NHEAD), 256, 0, stream>>>(qb, kb, vb, attnb);
    gemm(attnb, Wo + (size_t)i * D_MODEL * D_MODEL, qb, T_TOK, D_MODEL, D_MODEL); // qb reused
    resadd_rms_kernel<<<T_TOK, 256, 0, stream>>>(x, qb, post_attn + (size_t)i * D_MODEL);

    // GeGLU MLP sub-block
    rmsnorm_kernel<<<T_TOK, 256, 0, stream>>>(x, pre_ffw + (size_t)i * D_MODEL, hx);
    gemm(hx, Wg + (size_t)i * D_MODEL * FF_DIM, g, T_TOK, FF_DIM, D_MODEL);
    gemm(hx, Wu + (size_t)i * D_MODEL * FF_DIM, u, T_TOK, FF_DIM, D_MODEL);
    geglu_kernel<<<(T_TOK * FF_DIM + 255) / 256, 256, 0, stream>>>(g, u);
    gemm(g, Wd + (size_t)i * FF_DIM * D_MODEL, attnb, T_TOK, D_MODEL, FF_DIM); // attnb reused
    resadd_rms_kernel<<<T_TOK, 256, 0, stream>>>(x, attnb, post_ffw + (size_t)i * D_MODEL);
  }

  finalize_kernel<<<(T_TOK * D_MODEL + 255) / 256, 256, 0, stream>>>(x, std_bias, std_scale, out);
}
